// Multiscale_Block_81295140978727
// MI455X (gfx1250) — compile-verified, matrix operand order checked by reading, not run on hardware
//
#include <hip/hip_runtime.h>
#include <hip/hip_bf16.h>
#include <cstdint>

typedef __attribute__((ext_vector_type(16))) __bf16 v16bf;
typedef __attribute__((ext_vector_type(8)))  float  v8f;

#define DIMC   768
#define HEADS  12
#define NCLS   313
#define TT     377
#define TPAD   384
#define BATCH  4
#define NTOK   569
#define MLPD   3072

__device__ __forceinline__ unsigned short f2bf(float f){
  union { float f; unsigned u; } a; a.f = f;
  unsigned u = a.u;
  u += 0x7FFFu + ((u >> 16) & 1u);
  return (unsigned short)(u >> 16);
}

// WMMA 16x16x32 bf16 operand fragment loader.
// A-layout (ISA 7.12.2): lane L holds row M=L&15; halves 0..7 = K[h*8..h*8+7],
// halves 8..15 = K[16+h*8 .. 23+h*8] where h = L>>4.  Same pattern serves the
// B operand when the B tile is stored K-transposed ([N][K]) in LDS.
__device__ __forceinline__ v16bf load_frag(const unsigned short* lds, int row, int stride,
                                           int kbase, int lane){
  const int h = (lane >> 4) & 1;
  const unsigned short* p = lds + row * stride + kbase + h * 8;
  union { v16bf v; uint4 q[2]; } u;
  u.q[0] = *(const uint4*)(p);
  u.q[1] = *(const uint4*)(p + 16);
  return u.v;
}

__device__ __forceinline__ float block_sum_256(float v){
  __shared__ float red[8];
  #pragma unroll
  for (int o = 16; o > 0; o >>= 1) v += __shfl_xor(v, o, 32);
  const int wid = threadIdx.x >> 5, lane = threadIdx.x & 31;
  if (lane == 0) red[wid] = v;
  __syncthreads();
  float r = (lane < 8) ? red[lane] : 0.f;
  #pragma unroll
  for (int o = 4; o > 0; o >>= 1) r += __shfl_xor(r, o, 32);
  r = __shfl(r, 0, 32);
  __syncthreads();
  return r;
}

// ---------------- fp32 -> bf16 weight conversion ----------------
__global__ void cvt_bf16_kernel(const float* __restrict__ s, unsigned short* __restrict__ d, int n){
  int i = blockIdx.x * 256 + threadIdx.x;
  if (i < n) d[i] = f2bf(s[i]);
}

// ---------------- mask reduction: (B,65536,313) -> mw[B][4][64][313] ----------------
// One workgroup per (b, 16x16 grid cell); one lane per class => coalesced streams.
__global__ __launch_bounds__(320) void mask_reduce_kernel(const float* __restrict__ mask,
                                                          float* __restrict__ mw){
  const int cell = blockIdx.x;
  const int b = cell >> 8, rem = cell & 255;
  const int gy = rem >> 4, gx = rem & 15;
  const int cls = threadIdx.x;
  if (cls >= NCLS) return;
  const float* base = mask + (size_t)b * 65536 * NCLS;
  float s = 0.f;
  #pragma unroll 4
  for (int py = 0; py < 16; ++py){
    const float* rp = base + ((size_t)(gy * 16 + py) * 256 + gx * 16) * NCLS + cls;
    #pragma unroll 4
    for (int px = 0; px < 16; ++px) s += rp[(size_t)px * NCLS];
  }
  const int w = (gy >> 3) * 2 + (gx >> 3);
  const int pos = (gy & 7) * 8 + (gx & 7);
  mw[((size_t)((b * 4 + w) * 64 + pos)) * NCLS + cls] = s;
}

// ---------------- LN1: x -> window-ordered patch tokens (bf16) + class tokens (bf16) ----------------
__global__ __launch_bounds__(256) void ln1_kernel(const float* __restrict__ x,
    const float* __restrict__ g, const float* __restrict__ be,
    unsigned short* __restrict__ xnw, unsigned short* __restrict__ ctbf){
  const int n = blockIdx.x, b = blockIdx.y;
  const float* row = x + (size_t)(b * NTOK + n) * DIMC;
  float v[3];
  #pragma unroll
  for (int i = 0; i < 3; ++i) v[i] = row[threadIdx.x + i * 256];
  const float mean = block_sum_256(v[0] + v[1] + v[2]) * (1.f / DIMC);
  float sq = 0.f;
  #pragma unroll
  for (int i = 0; i < 3; ++i){ float d = v[i] - mean; sq += d * d; }
  const float var = block_sum_256(sq) * (1.f / DIMC);
  const float inv = rsqrtf(var + 1e-5f);
  unsigned short* dst;
  if (n < 256){
    const int rw = n >> 4, cl = n & 15;
    const int w = (rw >> 3) * 2 + (cl >> 3), pos = (rw & 7) * 8 + (cl & 7);
    dst = xnw + (size_t)((b * 4 + w) * 64 + pos) * DIMC;
  } else {
    dst = ctbf + (size_t)(b * NCLS + (n - 256)) * DIMC;
  }
  #pragma unroll
  for (int i = 0; i < 3; ++i){
    const int c = threadIdx.x + i * 256;
    dst[c] = f2bf((v[i] - mean) * inv * g[c] + be[c]);
  }
}

// ---------------- residual + LN2 (fused) ----------------
__global__ __launch_bounds__(256) void resid_ln2_kernel(const float* __restrict__ x,
    const float* __restrict__ patch, const float* __restrict__ ctf,
    const float* __restrict__ g, const float* __restrict__ be,
    float* __restrict__ xres, unsigned short* __restrict__ h1){
  const int n = blockIdx.x, b = blockIdx.y;
  const size_t ro = (size_t)(b * NTOK + n) * DIMC;
  const float* src;
  if (n < 256){
    const int rw = n >> 4, cl = n & 15;
    const int w = (rw >> 3) * 2 + (cl >> 3), pos = (rw & 7) * 8 + (cl & 7);
    src = patch + (size_t)((b * 4 + w) * 64 + pos) * DIMC;
  } else {
    src = ctf + (size_t)(b * NCLS + (n - 256)) * DIMC;
  }
  float v[3];
  #pragma unroll
  for (int i = 0; i < 3; ++i){
    const int c = threadIdx.x + i * 256;
    v[i] = x[ro + c] + src[c];
    xres[ro + c] = v[i];
  }
  const float mean = block_sum_256(v[0] + v[1] + v[2]) * (1.f / DIMC);
  float sq = 0.f;
  #pragma unroll
  for (int i = 0; i < 3; ++i){ float d = v[i] - mean; sq += d * d; }
  const float var = block_sum_256(sq) * (1.f / DIMC);
  const float inv = rsqrtf(var + 1e-5f);
  #pragma unroll
  for (int i = 0; i < 3; ++i){
    const int c = threadIdx.x + i * 256;
    h1[ro + c] = f2bf((v[i] - mean) * inv * g[c] + be[c]);
  }
}

// ---------------- generic bf16 WMMA GEMM with mode-specific A-gather / epilogue ----------------
struct GemmP {
  const unsigned short* A;      // generic A (bf16)
  const unsigned short* xnw;    // MODE 0: window-ordered patch tokens
  const unsigned short* ctbf;   // MODE 0: class-token carry (bf16)
  const unsigned short* Bw;     // weights bf16, row-major K x N
  const float* bias;
  unsigned short* outQ; unsigned short* outK; unsigned short* outV;   // MODE 0
  float* patch; float* ctf; unsigned short* ctbfOut;                  // MODE 1
  unsigned short* outBF;                                              // MODE 2
  const float* resid; float* outF;                                    // MODE 3
  int M, N, K, win;
};

template<int MODE>
__global__ __launch_bounds__(256) void gemm_bf16(GemmP p){
  __shared__ unsigned short As[128 * 40];   // [row][k], padded stride 40
  __shared__ unsigned short Bs[128 * 40];   // K-transposed: [n][k]
  const int tid = threadIdx.x;
  const int lane = tid & 31, wid = tid >> 5;
  const int waveM = wid & 3, waveN = wid >> 2;     // 4x2 wave grid -> wave tile 32x64
  const int bm = blockIdx.y * 128, bn = blockIdx.x * 128;
  v8f acc[2][4] = {};

  const int arow = tid >> 1, aseg = (tid & 1) * 16;       // A: 128 rows x 32 halves
  const int bk = tid >> 3, bnseg = (tid & 7) * 16;        // B: 32 k-rows x 128 cols
  const int gr = bm + arow;
  const unsigned short* aptr = nullptr;
  if (gr < p.M){
    if constexpr (MODE == 0){
      const int bb = gr / TT, t = gr - bb * TT;
      aptr = (t < 64) ? p.xnw + (size_t)((bb * 4 + p.win) * 64 + t) * DIMC
                      : p.ctbf + (size_t)(bb * NCLS + (t - 64)) * DIMC;
    } else {
      aptr = p.A + (size_t)gr * p.K;
    }
  }
  const unsigned short* bptr = p.Bw + (size_t)bk * p.N + bn + bnseg;

  for (int k0 = 0; k0 < p.K; k0 += 32){
    __syncthreads();
    uint4 a0, a1;
    if (aptr){
      const uint4* ap = (const uint4*)(aptr + k0 + aseg);
      a0 = ap[0]; a1 = ap[1];
    } else {
      a0 = make_uint4(0u, 0u, 0u, 0u); a1 = a0;
    }
    *(uint4*)&As[arow * 40 + aseg]     = a0;
    *(uint4*)&As[arow * 40 + aseg + 8] = a1;

    union { uint4 q[2]; unsigned short h[16]; } ub;
    const uint4* bp = (const uint4*)(bptr + (size_t)k0 * p.N);
    ub.q[0] = bp[0]; ub.q[1] = bp[1];
    #pragma unroll
    for (int i = 0; i < 16; ++i) Bs[(bnseg + i) * 40 + bk] = ub.h[i];

    if (k0 + 32 < p.K){
      if (aptr) __builtin_prefetch(aptr + k0 + 32 + aseg, 0, 0);
      __builtin_prefetch(bptr + (size_t)(k0 + 32) * p.N, 0, 0);
    }
    __syncthreads();

    v16bf af[2], bfv[4];
    #pragma unroll
    for (int mt = 0; mt < 2; ++mt)
      af[mt] = load_frag(As, waveM * 32 + mt * 16 + (lane & 15), 40, 0, lane);
    #pragma unroll
    for (int nt = 0; nt < 4; ++nt)
      bfv[nt] = load_frag(Bs, waveN * 64 + nt * 16 + (lane & 15), 40, 0, lane);
    #pragma unroll
    for (int mt = 0; mt < 2; ++mt)
      #pragma unroll
      for (int nt = 0; nt < 4; ++nt)
        acc[mt][nt] = __builtin_amdgcn_wmma_f32_16x16x32_bf16(
            false, af[mt], false, bfv[nt], (short)0, acc[mt][nt], false, false);
  }

  const int h2 = (lane >> 4) & 1, ln = lane & 15;
  #pragma unroll
  for (int mt = 0; mt < 2; ++mt){
    #pragma unroll
    for (int nt = 0; nt < 4; ++nt){
      #pragma unroll
      for (int r = 0; r < 8; ++r){
        const int gM = bm + waveM * 32 + mt * 16 + h2 * 8 + r;
        const int gN = bn + waveN * 64 + nt * 16 + ln;
        if (gM >= p.M) continue;
        const float val = acc[mt][nt][r] + p.bias[gN];
        if constexpr (MODE == 0){            // QKV scatter, head-major bf16
          const int part = gN / DIMC, rem = gN - part * DIMC;
          const int hh = rem >> 6, d = rem & 63;
          const int bb = gM / TT, t = gM - bb * TT;
          unsigned short* dst = (part == 0) ? p.outQ : ((part == 1) ? p.outK : p.outV);
          dst[((size_t)(bb * HEADS + hh) * TPAD + t) * 64 + d] = f2bf(val);
        } else if constexpr (MODE == 1){     // proj scatter: patch vs class-token carry
          const int bb = gM / TT, t = gM - bb * TT;
          if (t < 64){
            p.patch[(size_t)((bb * 4 + p.win) * 64 + t) * DIMC + gN] = val;
          } else {
            const size_t ci = (size_t)(bb * NCLS + (t - 64)) * DIMC + gN;
            p.ctf[ci] = val;
            p.ctbfOut[ci] = f2bf(val);
          }
        } else if constexpr (MODE == 2){     // FC1 + exact GELU
          const float ge = 0.5f * val * (1.0f + erff(val * 0.70710678f));
          p.outBF[(size_t)gM * MLPD + gN] = f2bf(ge);
        } else {                             // FC2 + residual
          p.outF[(size_t)gM * DIMC + gN] = val + p.resid[(size_t)gM * DIMC + gN];
        }
      }
    }
  }
}

// ---------------- fused masked flash-attention: one WG per (b, head, 128-query tile) ----------------
__global__ __launch_bounds__(256) void attn_kernel(const unsigned short* __restrict__ qb,
    const unsigned short* __restrict__ kb, const unsigned short* __restrict__ vb,
    const float* __restrict__ mw, unsigned short* __restrict__ obf, int win){
  __shared__ unsigned short Qs[128 * 72];   // [qrow][d]
  __shared__ unsigned short Ks[64 * 72];    // [key][d]   (== B^T layout for S = Q K^T)
  __shared__ unsigned short Vt[64 * 72];    // [d][key]   (== B^T layout for P V)
  __shared__ unsigned short Ps[128 * 72];   // staged P, per-wave 16-row slice

  const int tid = threadIdx.x, lane = tid & 31, wv = tid >> 5;
  const int qblk = blockIdx.x, h = blockIdx.y, b = blockIdx.z;
  const size_t headbase = (size_t)(b * HEADS + h) * TPAD * 64;
  const int h2 = (lane >> 4) & 1, ln = lane & 15;
  const float scale = 0.125f;                // hd^-0.5, hd=64
  const float* mwb = mw + (size_t)(b * 4 + win) * 64 * NCLS;

  { // load 128x64 Q tile
    const int row = tid >> 1, seg = (tid & 1) * 32;
    const uint4* sp = (const uint4*)(qb + headbase + (size_t)(qblk * 128 + row) * 64 + seg);
    uint4* dp = (uint4*)&Qs[row * 72 + seg];
    dp[0] = sp[0]; dp[1] = sp[1]; dp[2] = sp[2]; dp[3] = sp[3];
  }

  float mrun[8], lrun[8];
  v8f Oacc[4] = {};
  #pragma unroll
  for (int r = 0; r < 8; ++r){ mrun[r] = -3.0e38f; lrun[r] = 0.f; }

  for (int kt = 0; kt < 6; ++kt){
    __syncthreads();
    if (tid < 128){            // K tile 64x64
      const int row = tid >> 1, seg = (tid & 1) * 32;
      const uint4* sp = (const uint4*)(kb + headbase + (size_t)(kt * 64 + row) * 64 + seg);
      uint4* dp = (uint4*)&Ks[row * 72 + seg];
      dp[0] = sp[0]; dp[1] = sp[1]; dp[2] = sp[2]; dp[3] = sp[3];
    } else {                   // V tile transposed into Vt[d][key]
      const int t2 = tid - 128;
      const int key = t2 >> 1, seg = (t2 & 1) * 32;
      union { uint4 q[4]; unsigned short hh[32]; } uv;
      const uint4* sp = (const uint4*)(vb + headbase + (size_t)(kt * 64 + key) * 64 + seg);
      uv.q[0] = sp[0]; uv.q[1] = sp[1]; uv.q[2] = sp[2]; uv.q[3] = sp[3];
      #pragma unroll
      for (int i = 0; i < 32; ++i) Vt[(seg + i) * 72 + key] = uv.hh[i];
    }
    __syncthreads();

    // S tile (16 rows per wave x 64 keys)
    v8f st[4];
    #pragma unroll
    for (int nt = 0; nt < 4; ++nt){
      v8f s = {};
      #pragma unroll
      for (int kk = 0; kk < 64; kk += 32){
        v16bf qa = load_frag(Qs, wv * 16 + ln, 72, kk, lane);
        v16bf kf = load_frag(Ks, nt * 16 + ln, 72, kk, lane);
        s = __builtin_amdgcn_wmma_f32_16x16x32_bf16(false, qa, false, kf, (short)0, s, false, false);
      }
      st[nt] = s;
    }

    // mask + scale
    #pragma unroll
    for (int nt = 0; nt < 4; ++nt){
      const int kj = kt * 64 + nt * 16 + ln;
      #pragma unroll
      for (int r = 0; r < 8; ++r){
        const int qi = qblk * 128 + wv * 16 + h2 * 8 + r;
        bool ok;
        if (kj >= TT)                       ok = false;
        else if (qi >= TT)                  ok = true;      // padded rows, never stored
        else if ((qi < 64) == (kj < 64))    ok = true;      // patch-patch / cls-cls
        else if (qi < 64)                   ok = mwb[qi * NCLS + (kj - 64)] != 0.f;
        else                                ok = mwb[kj * NCLS + (qi - 64)] != 0.f;
        st[nt][r] = ok ? st[nt][r] * scale : -1e30f;
      }
    }

    // online softmax (rows live in half-waves -> width-16 reductions)
    #pragma unroll
    for (int r = 0; r < 8; ++r){
      float tmax = fmaxf(fmaxf(st[0][r], st[1][r]), fmaxf(st[2][r], st[3][r]));
      #pragma unroll
      for (int o = 1; o < 16; o <<= 1) tmax = fmaxf(tmax, __shfl_xor(tmax, o, 16));
      const float nm = fmaxf(mrun[r], tmax);
      const float fct = __expf(mrun[r] - nm);
      mrun[r] = nm;
      float rs = 0.f;
      #pragma unroll
      for (int nt = 0; nt < 4; ++nt){
        const float pv = (st[nt][r] <= -5e29f) ? 0.f : __expf(st[nt][r] - nm);
        st[nt][r] = pv;
        rs += pv;
      }
      #pragma unroll
      for (int o = 1; o < 16; o <<= 1) rs += __shfl_xor(rs, o, 16);
      lrun[r] = lrun[r] * fct + rs;
      #pragma unroll
      for (int ot = 0; ot < 4; ++ot) Oacc[ot][r] *= fct;
    }

    // stage P (bf16) through this wave's private LDS slice to re-layout as A-fragments
    #pragma unroll
    for (int nt = 0; nt < 4; ++nt)
      #pragma unroll
      for (int r = 0; r < 8; ++r)
        Ps[(wv * 16 + h2 * 8 + r) * 72 + nt * 16 + ln] = f2bf(st[nt][r]);

    // O += P @ V
    #pragma unroll
    for (int ot = 0; ot < 4; ++ot){
      #pragma unroll
      for (int kk = 0; kk < 64; kk += 32){
        v16bf pf = load_frag(Ps, wv * 16 + ln, 72, kk, lane);
        v16bf vf = load_frag(Vt, ot * 16 + ln, 72, kk, lane);
        Oacc[ot] = __builtin_amdgcn_wmma_f32_16x16x32_bf16(false, pf, false, vf, (short)0, Oacc[ot], false, false);
      }
    }
  }

  // finalize: O / l, write bf16 in (b, t, h*64+d) order (contiguous rows for proj GEMM)
  #pragma unroll
  for (int r = 0; r < 8; ++r){
    const int qi = qblk * 128 + wv * 16 + h2 * 8 + r;
    if (qi >= TT) continue;
    const float invl = 1.f / lrun[r];
    #pragma unroll
    for (int ot = 0; ot < 4; ++ot)
      obf[(size_t)(b * TT + qi) * DIMC + h * 64 + ot * 16 + ln] = f2bf(Oacc[ot][r] * invl);
  }
}

extern "C" void kernel_launch(void* const* d_in, const int* in_sizes, int n_in,
                              void* d_out, int out_size, void* d_ws, size_t ws_size,
                              hipStream_t stream){
  (void)in_sizes; (void)n_in; (void)out_size; (void)ws_size;
  const float* x     = (const float*)d_in[0];
  const float* mask  = (const float*)d_in[1];
  const float* ln1g  = (const float*)d_in[2];
  const float* ln1b  = (const float*)d_in[3];
  const float* qkvw  = (const float*)d_in[4];
  const float* qkvb  = (const float*)d_in[5];
  const float* projw = (const float*)d_in[6];
  const float* projb = (const float*)d_in[7];
  const float* ln2g  = (const float*)d_in[8];
  const float* ln2b  = (const float*)d_in[9];
  const float* fc1w  = (const float*)d_in[10];
  const float* fc1b  = (const float*)d_in[11];
  const float* fc2w  = (const float*)d_in[12];
  const float* fc2b  = (const float*)d_in[13];
  float* out = (float*)d_out;

  char* base = (char*)d_ws;
  size_t off = 0;
  auto alloc = [&](size_t bytes) -> void* {
    void* p = base + off;
    off = (off + bytes + 255) & ~(size_t)255;
    return p;
  };
  unsigned short* wqkv  = (unsigned short*)alloc((size_t)DIMC * 3 * DIMC * 2);
  unsigned short* wproj = (unsigned short*)alloc((size_t)DIMC * DIMC * 2);
  unsigned short* wfc1  = (unsigned short*)alloc((size_t)DIMC * MLPD * 2);
  unsigned short* wfc2  = (unsigned short*)alloc((size_t)MLPD * DIMC * 2);
  float*          mw    = (float*)alloc((size_t)BATCH * 4 * 64 * NCLS * 4);
  unsigned short* xnw   = (unsigned short*)alloc((size_t)BATCH * 4 * 64 * DIMC * 2);
  unsigned short* ctbf  = (unsigned short*)alloc((size_t)BATCH * NCLS * DIMC * 2);
  float*          ctf   = (float*)alloc((size_t)BATCH * NCLS * DIMC * 4);
  unsigned short* qbuf  = (unsigned short*)alloc((size_t)BATCH * HEADS * TPAD * 64 * 2);
  unsigned short* kbuf  = (unsigned short*)alloc((size_t)BATCH * HEADS * TPAD * 64 * 2);
  unsigned short* vbuf  = (unsigned short*)alloc((size_t)BATCH * HEADS * TPAD * 64 * 2);
  unsigned short* obf   = (unsigned short*)alloc((size_t)BATCH * TT * DIMC * 2);
  float*          patch = (float*)alloc((size_t)BATCH * 4 * 64 * DIMC * 4);
  float*          xres  = (float*)alloc((size_t)BATCH * NTOK * DIMC * 4);
  unsigned short* h1    = (unsigned short*)alloc((size_t)BATCH * NTOK * DIMC * 2);
  unsigned short* h2b   = (unsigned short*)alloc((size_t)BATCH * NTOK * MLPD * 2);

  auto cvt = [&](const float* s, unsigned short* d, int n){
    cvt_bf16_kernel<<<(n + 255) / 256, 256, 0, stream>>>(s, d, n);
  };
  cvt(qkvw,  wqkv,  DIMC * 3 * DIMC);
  cvt(projw, wproj, DIMC * DIMC);
  cvt(fc1w,  wfc1,  DIMC * MLPD);
  cvt(fc2w,  wfc2,  MLPD * DIMC);

  mask_reduce_kernel<<<BATCH * 256, 320, 0, stream>>>(mask, mw);
  ln1_kernel<<<dim3(NTOK, BATCH), 256, 0, stream>>>(x, ln1g, ln1b, xnw, ctbf);

  for (int w = 0; w < 4; ++w){
    GemmP pq = {};
    pq.xnw = xnw; pq.ctbf = ctbf; pq.Bw = wqkv; pq.bias = qkvb;
    pq.outQ = qbuf; pq.outK = kbuf; pq.outV = vbuf;
    pq.M = BATCH * TT; pq.N = 3 * DIMC; pq.K = DIMC; pq.win = w;
    gemm_bf16<0><<<dim3(18, 12), 256, 0, stream>>>(pq);

    attn_kernel<<<dim3(3, HEADS, BATCH), 256, 0, stream>>>(qbuf, kbuf, vbuf, mw, obf, w);

    GemmP pp = {};
    pp.A = obf; pp.Bw = wproj; pp.bias = projb;
    pp.patch = patch; pp.ctf = ctf; pp.ctbfOut = ctbf;
    pp.M = BATCH * TT; pp.N = DIMC; pp.K = DIMC; pp.win = w;
    gemm_bf16<1><<<dim3(6, 12), 256, 0, stream>>>(pp);
  }

  resid_ln2_kernel<<<dim3(NTOK, BATCH), 256, 0, stream>>>(x, patch, ctf, ln2g, ln2b, xres, h1);

  GemmP p1 = {};
  p1.A = h1; p1.Bw = wfc1; p1.bias = fc1b; p1.outBF = h2b;
  p1.M = BATCH * NTOK; p1.N = MLPD; p1.K = DIMC;
  gemm_bf16<2><<<dim3(24, 18), 256, 0, stream>>>(p1);

  GemmP p2 = {};
  p2.A = h2b; p2.Bw = wfc2; p2.bias = fc2b; p2.resid = xres; p2.outF = out;
  p2.M = BATCH * NTOK; p2.N = DIMC; p2.K = MLPD;
  gemm_bf16<3><<<dim3(6, 18), 256, 0, stream>>>(p2);
}
